// RNN_39633958208033
// MI455X (gfx1250) — matrix operand order repaired and checked by reading, not compiled, not run on hardware
//
#include <hip/hip_runtime.h>
#include <hip/hip_bf16.h>

// ---------------------------------------------------------------------------
// LSTM (H=51, 4H=204) over T=999 sequential steps, batch 4096, fused FC(51->1).
// Latency-chain bound: runtime ~= 999 * per-step latency. Everything on-chip:
// W_hh/W_fc resident in VGPRs as f16 WMMA B-operand (64 VGPRs/wave -> no
// spills), h in LDS (f16), c in registers, gates staged through LDS. The FC
// output is produced as an extra B column (N=204) of the recurrent WMMA.
// 4 waves/WG = 1 wave per SIMD32 of the WGP; 16 batch rows per WG; 256 WGs.
// ---------------------------------------------------------------------------

typedef __attribute__((ext_vector_type(16))) _Float16      v16h;
typedef __attribute__((ext_vector_type(8)))  float         v8f;
typedef __attribute__((ext_vector_type(4)))  float         v4f;
typedef __attribute__((ext_vector_type(4)))  unsigned int  v4u;

#define HID   51
#define GATES 204          // 4*HID
#define BATCH 4096
#define SEQ   999
#define HP    64           // K padded to 64 (2 x K=32 WMMA chunks)
#define NT    16           // N tiles: 13 real (col 204 = W_fc), 3 zero-dummy
#define GP    (NT * 16)    // 256 padded gate columns
#define ROWS  16           // batch rows per workgroup (one WMMA M tile)
#define WAVES 4
#define TPW   (NT / WAVES) // 4 N-tiles per wave
#define TPB   (WAVES * 32) // 128 threads
#define EWIT  7            // per-lane elementwise iters: k = kbase + 8*j < 51

__device__ __forceinline__ float fsig(float x) {
    // sigmoid(x) = 1 / (1 + 2^(-x*log2(e)))  -> v_exp_f32 + v_rcp_f32
    float t = __builtin_amdgcn_exp2f(x * -1.4426950408889634f);
    return __builtin_amdgcn_rcpf(1.0f + t);
}

__device__ __forceinline__ float ftanh(float x) {
#if defined(__has_builtin) && __has_builtin(__builtin_amdgcn_tanhf)
    return __builtin_amdgcn_tanhf(x);   // v_tanh_f32 (CDNA5 trans op)
#else
    return 2.0f * fsig(2.0f * x) - 1.0f;
#endif
}

__device__ __forceinline__ v8f wmma_f16(v16h a, v16h b, v8f c) {
    return __builtin_amdgcn_wmma_f32_16x16x32_f16(
        /*neg_a=*/false, a, /*neg_b=*/false, b,
        /*c_mod=*/(short)0, c, /*reuse_a=*/false, /*reuse_b=*/false);
}

__global__ __launch_bounds__(TPB)
void lstm_wmma_kernel(const float* __restrict__ input,   // (B, T, 1)
                      const float* __restrict__ W_ih,    // (4H, 1)
                      const float* __restrict__ W_hh,    // (4H, H)
                      const float* __restrict__ b_ih,    // (4H)
                      const float* __restrict__ b_hh,    // (4H)
                      const float* __restrict__ W_fc,    // (1, H)
                      const float* __restrict__ b_fc,    // (1)
                      float* __restrict__ out)           // (B, T, 1)
{
    __shared__ __align__(16) float    gbuf[GP * 16];    // gates, [n][m] (16 KB)
    __shared__ __align__(16) _Float16 hbuf[ROWS * HP];  // h (f16), K-padded (2 KB)
    __shared__ float  xbuf[ROWS];
    __shared__ float2 wb[GATES];                        // {W_ih, b_ih+b_hh}
    __shared__ float  wfc[HID];

    const int tid     = threadIdx.x;
    const int lane    = tid & 31;
    const int w       = tid >> 5;           // wave id 0..3
    const int rowbase = blockIdx.x * ROWS;

    // ---- one-time setup -------------------------------------------------
    for (int i = tid; i < GATES; i += TPB) wb[i] = make_float2(W_ih[i], b_ih[i] + b_hh[i]);
    for (int i = tid; i < HID;   i += TPB) wfc[i] = W_fc[i];
    for (int i = tid; i < ROWS * HP; i += TPB) hbuf[i] = (_Float16)0.0f;
    const float bfc = b_fc[0];

    // B operand resident in VGPRs: 4 N-tiles x 2 K-chunks per wave (64 VGPRs).
    // 16-bit B (32x16, K x N) layout: lanes 0-15 -> N=lane, K=0..7,16..23;
    //                                 lanes 16-31 -> N=lane-16, K=8..15,24..31.
    v16h bw[TPW][2];
    {
        const int nl    = lane & 15;
        const int khalf = (lane < 16) ? 0 : 16;
        for (int ti = 0; ti < TPW; ++ti) {
            const int n = (w * TPW + ti) * 16 + nl;     // padded gate column
            for (int kc = 0; kc < 2; ++kc) {
                union { v16h v; _Float16 e[16]; } u;
                for (int q = 0; q < 16; ++q) {
                    const int k = kc * 32 + (q < 8 ? (khalf >> 1) + q : 16 + (khalf >> 1) + (q - 8));
                    float val = 0.0f;
                    if (k < HID) {
                        if (n < GATES)       val = W_hh[n * HID + k];
                        else if (n == GATES) val = W_fc[k];      // fused FC column
                    }
                    u.e[q] = (_Float16)val;
                }
                bw[ti][kc] = u.v;
            }
        }
    }

    float c[EWIT];
    #pragma unroll
    for (int j = 0; j < EWIT; ++j) c[j] = 0.0f;

    __syncthreads();

    const int m     = lane & 15;
    const int kb    = (lane < 16) ? 0 : 8;  // 16-bit A layout K sub-base
    const int mb    = (lane < 16) ? 0 : 8;  // f32 D layout M sub-base
    const int mm    = tid & 15;             // elementwise row (constant: 128%16==0)
    const int kbase = tid >> 4;             // elementwise k base, 0..7

    // Software-pipelined input: xv holds x_t while x_{t+1} loads during step t.
    float xv = 0.0f;
    if (tid < ROWS) xv = input[(size_t)(rowbase + tid) * SEQ];

    // ---- sequential time loop ------------------------------------------
    for (int t = 0; t < SEQ; ++t) {
        if (tid < ROWS) xbuf[tid] = xv;     // x_t -> LDS (covered by barrier 1)

        // A operand (h_{t-1}) from LDS. 16-bit A (16x32) layout:
        // lanes 0-15: M=lane, K = base+0..7 and base+16..23 (two b128 loads).
        v16h a0, a1;
        {
            const _Float16* hrow = &hbuf[m * HP];
            union { v16h v; v4u h[2]; } ua;
            ua.h[0] = *(const v4u*)&hrow[kb];
            ua.h[1] = *(const v4u*)&hrow[kb + 16];
            a0 = ua.v;
            ua.h[0] = *(const v4u*)&hrow[32 + kb];
            ua.h[1] = *(const v4u*)&hrow[32 + kb + 16];
            a1 = ua.v;
        }

        // gates(+y) = h_{t-1} @ [W_hh.T | W_fc.T] : 4 tiles/wave, K=64 each
        #pragma unroll
        for (int ti = 0; ti < TPW; ++ti) {
            v8f acc = {};
            acc = wmma_f16(a0, bw[ti][0], acc);
            acc = wmma_f16(a1, bw[ti][1], acc);
            const int n = (w * TPW + ti) * 16 + m;
            float* dst = &gbuf[n * 16 + mb];
            union { v8f v; v4f q[2]; } ud;
            ud.v = acc;
            *(v4f*)&dst[0] = ud.q[0];
            *(v4f*)&dst[4] = ud.q[1];
        }

        // Issue next step's input load now; waited only at next xbuf store.
        if (tid < ROWS && t + 1 < SEQ) {
            xv = input[(size_t)(rowbase + tid) * SEQ + (t + 1)];
            if (t + 32 < SEQ)
                __builtin_prefetch(&input[(size_t)(rowbase + tid) * SEQ + t + 32], 0, 1);
        }
        __syncthreads();

        // Fused FC output: gbuf column 204 holds h_{t-1} @ W_fc = y_{t-1}
        if (t > 0 && tid < ROWS)
            out[(size_t)(rowbase + tid) * SEQ + (t - 1)] = gbuf[GATES * 16 + tid] + bfc;

        // Elementwise LSTM cell: lane owns fixed row mm, units k = kbase+8j,
        // so c lives in registers for all 999 steps.
        const float x = xbuf[mm];
        #pragma unroll
        for (int j = 0; j < EWIT; ++j) {
            const int k = kbase + 8 * j;
            if (k < HID) {
                const float2 wbi = wb[k];
                const float2 wbf = wb[HID + k];
                const float2 wbg = wb[2 * HID + k];
                const float2 wbo = wb[3 * HID + k];
                const float gi = gbuf[(k          ) * 16 + mm] + x * wbi.x + wbi.y;
                const float gf = gbuf[(HID     + k) * 16 + mm] + x * wbf.x + wbf.y;
                const float gg = gbuf[(2 * HID + k) * 16 + mm] + x * wbg.x + wbg.y;
                const float go = gbuf[(3 * HID + k) * 16 + mm] + x * wbo.x + wbo.y;
                const float cn = fsig(gf) * c[j] + fsig(gi) * ftanh(gg);
                c[j] = cn;
                const float h  = fsig(go) * ftanh(cn);
                hbuf[mm * HP + k] = (_Float16)h;
            }
        }
        __syncthreads();   // h_t visible before next step's A loads
    }

    // ---- last timestep's FC output (h_{T-1} . W_fc + b) -----------------
    if (tid < ROWS) {
        float acc = bfc;
        for (int k = 0; k < HID; ++k)
            acc += (float)hbuf[tid * HP + k] * wfc[k];
        out[(size_t)(rowbase + tid) * SEQ + (SEQ - 1)] = acc;
    }
}

extern "C" void kernel_launch(void* const* d_in, const int* in_sizes, int n_in,
                              void* d_out, int out_size, void* d_ws, size_t ws_size,
                              hipStream_t stream) {
    (void)in_sizes; (void)n_in; (void)d_ws; (void)ws_size; (void)out_size;
    const float* input = (const float*)d_in[0];
    const float* W_ih  = (const float*)d_in[1];
    const float* W_hh  = (const float*)d_in[2];
    const float* b_ih  = (const float*)d_in[3];
    const float* b_hh  = (const float*)d_in[4];
    const float* W_fc  = (const float*)d_in[5];
    const float* b_fc  = (const float*)d_in[6];
    // d_in[7] = future (0 in this workload)
    float* out = (float*)d_out;

    dim3 grid(BATCH / ROWS);   // 256 workgroups
    dim3 block(TPB);           // 4 waves = 1 per SIMD32
    lstm_wmma_kernel<<<grid, block, 0, stream>>>(input, W_ih, W_hh, b_ih, b_hh,
                                                 W_fc, b_fc, out);
}
